// CARA1_38654705664770
// MI455X (gfx1250) — matrix operand order repaired and checked by reading, not compiled, not run on hardware
//
#include <hip/hip_runtime.h>
#include <hip/hip_bf16.h>

// ---------------------------------------------------------------------------
// CARA recurrence on MI455X (gfx1250, wave32).
//  - bf16 WMMA 16x16x32, fp32 accumulate (13 WMMA sites).
//  - Persistent 16-WG kernel, global generation barrier (4 syncs / step).
//  - Weights pre-packed to WMMA B-fragment lane order (bf16, L2-resident).
//  - Raw A-tile staging via GLOBAL_LOAD_ASYNC_TO_LDS_B32 (ASYNCcnt),
//    transformed staging (ua/un) via VALU + ds stores (DScnt).
// ---------------------------------------------------------------------------

#define OD      256
#define BATCH   128
#define TSTEPS  512
#define DIMS    (2*OD + 2)
#define NWG     16
#define THREADS 256
#define WAVES_PER_WG (THREADS/32)

typedef __attribute__((ext_vector_type(16))) __bf16         v16bf;
typedef __attribute__((ext_vector_type(16))) unsigned short v16u;
typedef __attribute__((ext_vector_type(8)))  float          v8f;

// packed weight slots (each OD*OD bf16 = 65536 elems). PK_AH,PK_AU must be
// adjacent: the a-gate GEMM runs K=512 across the concatenated pair.
enum { PK_WZ = 0, PK_WR, PK_WH, PK_WSUM, PK_WT, PK_WGG, PK_UZ, PK_UR, PK_UH,
       PK_AH, PK_AU, NPACK };
#define PACK_ELEMS (OD*OD)

// fp32 step buffers (each BATCH*OD = 32768 floats)
enum { S_A = 0, S_HZ, S_HR, S_XZ, S_XR, S_XH, S_T1, S_G1,
       S_UZ, S_UH, S_M, S_S, S_HU, S_H, NBUF };
#define BUFN (BATCH*OD)

__device__ __forceinline__ unsigned short f2bf(float f) {
    unsigned int u = __float_as_uint(f);
    u += 0x7FFFu + ((u >> 16) & 1u);           // round-to-nearest-even
    return (unsigned short)(u >> 16);
}

// ---------------------------------------------------------------------------
// Kernel 1: pack weights fp32 -> bf16 in WMMA B-fragment lane order.
// Layout per pack: elem[ kt*8192 + nt*512 + lane*16 + e ] = W[k][n]
//   k = kt*32 + (lane/16)*16 + e,  n = nt*16 + (lane%16)
// ---------------------------------------------------------------------------
__global__ void cara_pack(const float* __restrict__ Wz, const float* __restrict__ Wr,
                          const float* __restrict__ Wh, const float* __restrict__ Wt,
                          const float* __restrict__ Wg, const float* __restrict__ Uz,
                          const float* __restrict__ Ur, const float* __restrict__ Uh,
                          const float* __restrict__ Ah, const float* __restrict__ Au,
                          unsigned short* __restrict__ packs) {
    int unit = blockIdx.x;          // NPACK*8 units, one (pack, kt) each
    int p  = unit >> 3;
    int kt = unit & 7;
    const float* src;
    switch (p) {
        case PK_WZ:  src = Wz; break;  case PK_WR:  src = Wr; break;
        case PK_WH:  src = Wh; break;  case PK_WSUM: src = Wz; break;
        case PK_WT:  src = Wt; break;  case PK_WGG: src = Wg; break;
        case PK_UZ:  src = Uz; break;  case PK_UR:  src = Ur; break;
        case PK_UH:  src = Uh; break;  case PK_AH:  src = Ah; break;
        default:     src = Au; break;
    }
    unsigned short* dst = packs + (size_t)p * PACK_ELEMS + (size_t)kt * 8192;
    for (int i = threadIdx.x; i < 8192; i += blockDim.x) {
        int nt   = i >> 9;
        int rem  = i & 511;
        int lane = rem >> 4;
        int e    = rem & 15;
        int k = kt*32 + ((lane >> 4) << 4) + e;
        int n = nt*16 + (lane & 15);
        float v = src[k*OD + n];
        if (p == PK_WSUM) v = Wz[k*OD+n] + Wr[k*OD+n] + Wh[k*OD+n];
        dst[i] = f2bf(v);
    }
}

// ---------------------------------------------------------------------------
// Fragment helpers
// ---------------------------------------------------------------------------
__device__ __forceinline__ void lds_fence() {
    __builtin_amdgcn_wave_barrier();
    asm volatile("s_wait_dscnt 0" ::: "memory");   // CDNA5 split DS counter
    __builtin_amdgcn_wave_barrier();
}

// One per-lane async copy: 4B/lane, 128B/wave, global -> LDS, no VGPR data.
// VDST carries the workgroup-relative LDS byte offset (low 32 bits of the
// generic pointer); HW adds LDS_BASE. Tracked by ASYNCcnt.
__device__ __forceinline__ void async_row_copy(const float* gsrc, float* ldst) {
    unsigned lds_off = (unsigned)(uintptr_t)ldst;
    asm volatile("global_load_async_to_lds_b32 %0, %1, off"
                 :: "v"(lds_off), "v"(gsrc) : "memory");
}

__device__ __forceinline__ void async_fence() {
    asm volatile("s_wait_asynccnt 0" ::: "memory");
    __builtin_amdgcn_wave_barrier();
}

// A fragment (16x32 bf16) gather from a staged fp32 LDS tile (16 rows x 32 k,
// row stride 33 to spread banks). Lane l<16: row l, K {0..15};
// lane l>=16: row l-16, K {16..31} — matching the packed B order.
__device__ __forceinline__ v16bf a_frag(const float* sA, int lane) {
    v16u u;
    const float* row = sA + (lane & 15) * 33;
    int base = (lane & 16) ? 16 : 0;
#pragma unroll
    for (int e = 0; e < 16; ++e) {
        int kl = base + e;
        u[e] = f2bf(row[kl]);
    }
    return __builtin_bit_cast(v16bf, u);
}

// B fragment: contiguous 32B per lane from the packed weights.
__device__ __forceinline__ v16bf b_frag(const unsigned short* packs, int p,
                                        int kt, int nt, int lane) {
    const unsigned short* ptr =
        packs + (size_t)p * PACK_ELEMS + ((((size_t)kt*16 + nt)*32 + lane) << 4);
    return __builtin_bit_cast(v16bf, *(const v16u*)ptr);
}

#define WMMA(acc, a, b) \
    (acc) = __builtin_amdgcn_wmma_f32_16x16x32_bf16(false, (a), false, (b), \
                                                    (short)0, (acc), false, false)

// ---------------------------------------------------------------------------
// A-tile staging into LDS
// ---------------------------------------------------------------------------
// Raw copies: async global->LDS path. s_wait_dscnt 0 first so the previous
// fragment's ds gathers are complete before the tile is overwritten.
__device__ __forceinline__ void stage_h(float* sA, const float* src,
                                        int m0, int kb, int lane) {
    asm volatile("s_wait_dscnt 0" ::: "memory");
#pragma unroll
    for (int i = 0; i < 16; ++i)
        async_row_copy(&src[(m0 + i)*OD + kb + lane], &sA[i*33 + lane]);
    async_fence();
}

__device__ __forceinline__ void stage_xi(float* sA, const float* x, int t,
                                         int m0, int kb, int lane, int colBase) {
    asm volatile("s_wait_dscnt 0" ::: "memory");
#pragma unroll
    for (int i = 0; i < 16; ++i) {
        size_t idx = (size_t)(m0 + i) * ((size_t)TSTEPS * DIMS)
                   + (size_t)t * DIMS + colBase + kb + lane;
        async_row_copy(&x[idx], &sA[i*33 + lane]);
    }
    async_fence();
}

// ua = a*u  or  un = (1-a)*u, built on the fly (VALU transform -> ds stores;
// LDS ops are in-order per wave, so no extra guard vs. prior gathers needed)
__device__ __forceinline__ void stage_gate(float* sA, const float* Sa,
                                           const float* x, int t, int m0,
                                           int kb, int lane, bool is_un) {
#pragma unroll
    for (int i = 0; i < 16; ++i) {
        int b = m0 + i;
        float a = Sa[b*OD + kb + lane];
        float u = x[(size_t)b * ((size_t)TSTEPS * DIMS)
                    + (size_t)t * DIMS + OD + kb + lane];
        sA[i*33 + lane] = is_un ? (1.0f - a) * u : a * u;
    }
    lds_fence();
}

// ---------------------------------------------------------------------------
// Device-wide generation barrier (16 resident WGs)
// ---------------------------------------------------------------------------
__device__ __forceinline__ void grid_sync(int* cnt, int* gen) {
    __syncthreads();
    if (threadIdx.x == 0) {
        __threadfence();
        int g = __hip_atomic_load(gen, __ATOMIC_RELAXED, __HIP_MEMORY_SCOPE_AGENT);
        int prev = __hip_atomic_fetch_add(cnt, 1, __ATOMIC_ACQ_REL,
                                          __HIP_MEMORY_SCOPE_AGENT);
        if (prev == NWG - 1) {
            __hip_atomic_store(cnt, 0, __ATOMIC_RELAXED, __HIP_MEMORY_SCOPE_AGENT);
            __hip_atomic_fetch_add(gen, 1, __ATOMIC_RELEASE,
                                   __HIP_MEMORY_SCOPE_AGENT);
        } else {
            while (__hip_atomic_load(gen, __ATOMIC_ACQUIRE,
                                     __HIP_MEMORY_SCOPE_AGENT) == g)
                __builtin_amdgcn_s_sleep(2);
        }
    }
    __syncthreads();
}

// ---------------------------------------------------------------------------
// Kernel 2: persistent recurrence. 16 WGs x 8 waves = 128 wave slots
// == the 8x16 tile grid of every (128x256)x(256x256) GEMM.
// ---------------------------------------------------------------------------
#define BUF(i) (bufs + (size_t)(i) * BUFN)

__global__ __launch_bounds__(THREADS, 1)
void cara_persist(const float* __restrict__ x,
                  const float* __restrict__ bz, const float* __restrict__ br,
                  const float* __restrict__ bh, const float* __restrict__ bah,
                  const float* __restrict__ bau, const float* __restrict__ Ut,
                  const float* __restrict__ bt, const float* __restrict__ Ug,
                  const float* __restrict__ bg,
                  const unsigned short* __restrict__ packs,
                  float* __restrict__ bufs, int* __restrict__ bar,
                  float* __restrict__ out) {
    __shared__ float sAll[WAVES_PER_WG][16*33];
    const int lane = threadIdx.x & 31;
    const int wave = threadIdx.x >> 5;
    const int slot = blockIdx.x * WAVES_PER_WG + wave;   // 0..127
    const int bm = slot >> 4;                            // 0..7  (M tile)
    const int bn = slot & 15;                            // 0..15 (N tile)
    float* sA = sAll[wave];
    float* H  = BUF(S_H);
    int* cnt = bar;
    int* gen = bar + 1;
    const int gtid = blockIdx.x * THREADS + threadIdx.x;
    const int nlocal = bn*16 + (lane & 15);              // output column

    for (int t = 0; t < TSTEPS; ++t) {
        if (lane == 0 && t + 1 < TSTEPS)                 // global_prefetch_b8
            __builtin_prefetch(x + (size_t)(bm*16) * ((size_t)TSTEPS*DIMS)
                                 + (size_t)(t+1)*DIMS, 0, 1);

        // ---------------- stage 1: 8 GEMMs independent of this step -------
        v8f aXZ = {}, aXR = {}, aXH = {}, aT1 = {}, aG1 = {};
        for (int kt = 0; kt < 8; ++kt) {                 // A = xi
            stage_xi(sA, x, t, bm*16, kt*32, lane, 0);
            v16bf af = a_frag(sA, lane);
            WMMA(aXZ, af, b_frag(packs, PK_WZ,  kt, bn, lane));
            WMMA(aXR, af, b_frag(packs, PK_WR,  kt, bn, lane));
            WMMA(aXH, af, b_frag(packs, PK_WH,  kt, bn, lane));
            WMMA(aT1, af, b_frag(packs, PK_WT,  kt, bn, lane));
            WMMA(aG1, af, b_frag(packs, PK_WGG, kt, bn, lane));
        }
        v8f aHZ = {}, aHR = {};
        for (int kt = 0; kt < 8; ++kt) {                 // A = h
            stage_h(sA, H, bm*16, kt*32, lane);
            v16bf af = a_frag(sA, lane);
            WMMA(aHZ, af, b_frag(packs, PK_UZ, kt, bn, lane));
            WMMA(aHR, af, b_frag(packs, PK_UR, kt, bn, lane));
        }
        v8f aA = {};
        for (int kt = 0; kt < 16; ++kt) {                // A = [h | u], K=512
            if (kt < 8) stage_h(sA, H, bm*16, kt*32, lane);
            else        stage_xi(sA, x, t, bm*16, (kt-8)*32, lane, OD);
            v16bf af = a_frag(sA, lane);
            WMMA(aA, af, b_frag(packs, PK_AH, kt, bn, lane)); // spans PK_AU
        }
#pragma unroll
        for (int r = 0; r < 8; ++r) {
            int m   = bm*16 + r + ((lane & 16) ? 8 : 0);
            int idx = m*OD + nlocal;
            BUF(S_XZ)[idx] = aXZ[r];
            BUF(S_XR)[idx] = aXR[r];
            BUF(S_XH)[idx] = aXH[r];
            float tsc = x[(size_t)m*((size_t)TSTEPS*DIMS) + (size_t)t*DIMS + 2*OD];
            float gsc = x[(size_t)m*((size_t)TSTEPS*DIMS) + (size_t)t*DIMS + 2*OD + 1];
            BUF(S_T1)[idx] = tanhf(aT1[r] + tanhf(tsc*Ut[nlocal]) + bt[nlocal]);
            BUF(S_G1)[idx] = tanhf(aG1[r] + tanhf(gsc*Ug[nlocal]) + bg[nlocal]);
            BUF(S_HZ)[idx] = aHZ[r];
            BUF(S_HR)[idx] = aHR[r];
            BUF(S_A)[idx]  = tanhf(aA[r] + bah[nlocal] + bau[nlocal]);
        }
        grid_sync(cnt, gen);

        // ---------------- stage 2: gate GEMMs (need a) --------------------
        v8f aUZ = {}, aUR = {}, aUH = {}, aS = {};
        for (int kt = 0; kt < 8; ++kt) {
            stage_gate(sA, BUF(S_A), x, t, bm*16, kt*32, lane, false); // ua
            v16bf af = a_frag(sA, lane);
            WMMA(aUZ, af, b_frag(packs, PK_WZ, kt, bn, lane));
            WMMA(aUR, af, b_frag(packs, PK_WR, kt, bn, lane));
            WMMA(aUH, af, b_frag(packs, PK_WH, kt, bn, lane));
            stage_gate(sA, BUF(S_A), x, t, bm*16, kt*32, lane, true);  // un
            af = a_frag(sA, lane);
            WMMA(aS, af, b_frag(packs, PK_WSUM, kt, bn, lane));
        }
#pragma unroll
        for (int r = 0; r < 8; ++r) {
            int m   = bm*16 + r + ((lane & 16) ? 8 : 0);
            int idx = m*OD + nlocal;
            BUF(S_UZ)[idx] = aUZ[r];
            BUF(S_UH)[idx] = aUH[r];
            float rr = tanhf(BUF(S_XR)[idx] + BUF(S_HR)[idx] + aUR[r]
                             + 2.0f*br[nlocal]);
            BUF(S_M)[idx] = rr * BUF(S_T1)[idx] * BUF(S_G1)[idx] * H[idx];
            BUF(S_S)[idx] = 1.0f + aS[r] + bz[nlocal] + br[nlocal] + bh[nlocal];
        }
        grid_sync(cnt, gen);

        // ---------------- stage 3: hu = (r*t1*g1*h) @ U_h -----------------
        v8f aHU = {};
        for (int kt = 0; kt < 8; ++kt) {
            stage_h(sA, BUF(S_M), bm*16, kt*32, lane);
            v16bf af = a_frag(sA, lane);
            WMMA(aHU, af, b_frag(packs, PK_UH, kt, bn, lane));
        }
#pragma unroll
        for (int r = 0; r < 8; ++r) {
            int m = bm*16 + r + ((lane & 16) ? 8 : 0);
            BUF(S_HU)[m*OD + nlocal] = aHU[r];
        }
        grid_sync(cnt, gen);

        // ---------------- elementwise h update ----------------------------
        for (int i = gtid; i < BUFN; i += NWG*THREADS) {
            int n = i & (OD - 1);
            float z  = tanhf(BUF(S_XZ)[i] + BUF(S_HZ)[i] + BUF(S_UZ)[i]
                             + 2.0f*bz[n]);
            float hh = BUF(S_XH)[i] + BUF(S_HU)[i] + BUF(S_UH)[i] + 2.0f*bh[n];
            hh = fminf(fmaxf(hh * (1.0f/6.0f) + 0.5f, 0.0f), 1.0f);
            H[i] = (z*H[i] + (1.0f - z)*hh) * BUF(S_S)[i];
        }
        grid_sync(cnt, gen);
    }

    for (int i = gtid; i < BUFN; i += NWG*THREADS)
        out[i] = H[i];
}

// ---------------------------------------------------------------------------
extern "C" void kernel_launch(void* const* d_in, const int* in_sizes, int n_in,
                              void* d_out, int out_size, void* d_ws, size_t ws_size,
                              hipStream_t stream) {
    (void)in_sizes; (void)n_in; (void)out_size; (void)ws_size;
    const float* x   = (const float*)d_in[0];
    const float* Wz  = (const float*)d_in[1];
    const float* Uz  = (const float*)d_in[2];
    const float* bz  = (const float*)d_in[3];
    const float* Wr  = (const float*)d_in[4];
    const float* Ur  = (const float*)d_in[5];
    const float* br  = (const float*)d_in[6];
    const float* Wh  = (const float*)d_in[7];
    const float* Uh  = (const float*)d_in[8];
    const float* bh  = (const float*)d_in[9];
    const float* Ah  = (const float*)d_in[10];
    const float* Au  = (const float*)d_in[11];
    const float* bah = (const float*)d_in[12];
    const float* bau = (const float*)d_in[13];
    const float* Wt  = (const float*)d_in[14];
    const float* Ut  = (const float*)d_in[15];
    const float* bt  = (const float*)d_in[16];
    const float* Wg  = (const float*)d_in[17];
    const float* Ug  = (const float*)d_in[18];
    const float* bg  = (const float*)d_in[19];

    unsigned short* packs = (unsigned short*)d_ws;
    size_t packBytes = (size_t)NPACK * PACK_ELEMS * sizeof(unsigned short);
    float* bufs = (float*)((char*)d_ws + packBytes);
    size_t bufBytes = (size_t)NBUF * BUFN * sizeof(float);
    int* bar = (int*)((char*)d_ws + packBytes + bufBytes);

    hipMemsetAsync(bufs + (size_t)S_H * BUFN, 0, BUFN * sizeof(float), stream);
    hipMemsetAsync(bar, 0, 2 * sizeof(int), stream);

    cara_pack<<<NPACK*8, 256, 0, stream>>>(Wz, Wr, Wh, Wt, Wg, Uz, Ur, Uh,
                                           Ah, Au, packs);
    cara_persist<<<NWG, THREADS, 0, stream>>>(x, bz, br, bh, bah, bau, Ut, bt,
                                              Ug, bg, packs, bufs, bar,
                                              (float*)d_out);
}
